// MoE_5299989643592
// MI455X (gfx1250) — matrix operand order repaired and checked by reading, not compiled, not run on hardware
//
#include <hip/hip_runtime.h>
#include <hip/hip_bf16.h>

// MoE forward for MI455X (gfx1250, wave32, WMMA).
// Compute-bound at ~412 routed GFLOP -> bf16 WMMA (16x16x32) with f32 accum.
// Token-gather grouped GEMM per expert; experts sequential for determinism.
// Weights pre-converted to bf16 per expert; 32x256 block tiles; explicit
// register double-buffering; K passed at runtime so the hot loop stays a
// compact unroll-2 software pipeline (I-cache resident, bounded VGPRs).

constexpr int kT = 8192;
constexpr int kD = 2048;
constexpr int kH = 2048;
constexpr int kE = 8;

typedef __attribute__((ext_vector_type(16))) __bf16 v16bf;
typedef __attribute__((ext_vector_type(8)))  __bf16 v8bf;
typedef __attribute__((ext_vector_type(4)))  __bf16 v4bf;
typedef __attribute__((ext_vector_type(8)))  float  v8f;

// ---- WMMA helpers (CDNA5 wave32 layouts per CDNA5 ISA) ----

static __device__ __forceinline__ v8f wmma_bf16(v16bf a, v16bf b, v8f c) {
  // 8 args: (neg_a, A, neg_b, B, c_mod, C, reuse_a, reuse_b)
  return __builtin_amdgcn_wmma_f32_16x16x32_bf16(false, a, false, b, (short)0, c,
                                                 false, false);
}

// A 16x32 bf16 fragment. `ap` already points at row*ld + k + k0, where
// k0 = (lane>=16 ? 8 : 0). Elements 0..7 = K[k0..k0+8), 8..15 = K[16+k0..).
static __device__ __forceinline__ v16bf load_a_frag(const __bf16* __restrict__ ap) {
  const v8bf lo = *(const v8bf*)(ap);
  const v8bf hi = *(const v8bf*)(ap + 16);
  v16bf a;
#pragma unroll
  for (int i = 0; i < 8; ++i) { a[i] = lo[i]; a[i + 8] = hi[i]; }
  return a;
}

// B 32x16 bf16 fragment from row-major bf16 weights W[n, k] (k contiguous).
// `wp` already points at n*ld + k + koff, koff = (lane>=16 ? 8 : 0).
// Lane holds column n = lane%16; elements 0..7 = K[koff..koff+8), 8..15 = K[16+koff..).
static __device__ __forceinline__ v16bf load_b_frag(const __bf16* __restrict__ wp) {
  const v8bf lo = *(const v8bf*)(wp);
  const v8bf hi = *(const v8bf*)(wp + 16);
  v16bf b;
#pragma unroll
  for (int i = 0; i < 8; ++i) { b[i] = lo[i]; b[i + 8] = hi[i]; }
  return b;
}

// ---- small utility kernels ----

__global__ void moe_zero_f32(float* __restrict__ p, int n) {
  const int i = blockIdx.x * blockDim.x + threadIdx.x;
  if (i < n) p[i] = 0.0f;
}

__global__ void moe_zero_i32(int* __restrict__ p, int n) {
  const int i = blockIdx.x * blockDim.x + threadIdx.x;
  if (i < n) p[i] = 0;
}

__global__ void moe_cvt_bf16(const float* __restrict__ x, __bf16* __restrict__ y,
                             int n) {
  const int i = (blockIdx.x * blockDim.x + threadIdx.x) << 2;
  if (i < n) {
    const float4 f = *(const float4*)(x + i);
    v4bf o;
    o[0] = (__bf16)f.x; o[1] = (__bf16)f.y;
    o[2] = (__bf16)f.z; o[3] = (__bf16)f.w;
    *(v4bf*)(y + i) = o;
  }
}

// ---- router: one wave per token; softmax over 8 logits, top-2, append ----

__global__ void __launch_bounds__(256) moe_router(
    const float* __restrict__ X, const float* __restrict__ Gw,
    int* __restrict__ counts, int* __restrict__ idx, float* __restrict__ ce) {
  const int lane = threadIdx.x & 31;
  const int wave = threadIdx.x >> 5;
  const int t = (blockIdx.x << 3) + wave;  // 8 waves/block, one token each
  const float* xr = X + (size_t)t * kD;

  float p[kE];
#pragma unroll
  for (int e = 0; e < kE; ++e) p[e] = 0.0f;
  for (int d = lane; d < kD; d += 32) {
    const float xv = xr[d];
#pragma unroll
    for (int e = 0; e < kE; ++e) p[e] += xv * Gw[e * kD + d];
  }
#pragma unroll
  for (int e = 0; e < kE; ++e) {
#pragma unroll
    for (int off = 16; off > 0; off >>= 1) p[e] += __shfl_xor(p[e], off, 32);
  }

  if (lane == 0) {
    float mx = p[0];
#pragma unroll
    for (int e = 1; e < kE; ++e) mx = fmaxf(mx, p[e]);
    float s[kE], den = 0.0f;
#pragma unroll
    for (int e = 0; e < kE; ++e) { s[e] = __expf(p[e] - mx); den += s[e]; }
    const float inv = 1.0f / den;
    // top-2 on logits (softmax is monotone); earliest index wins ties
    int i1 = 0;
#pragma unroll
    for (int e = 1; e < kE; ++e) if (p[e] > p[i1]) i1 = e;
    int i2 = (i1 == 0) ? 1 : 0;
#pragma unroll
    for (int e = 0; e < kE; ++e) if (e != i1 && p[e] > p[i2]) i2 = e;

    int pos = atomicAdd(counts + i1, 1);
    idx[i1 * kT + pos] = t;
    ce[i1 * kT + pos]  = s[i1] * inv;
    pos = atomicAdd(counts + i2, 1);
    idx[i2 * kT + pos] = t;
    ce[i2 * kT + pos]  = s[i2] * inv;
  }
}

// ---- GEMM1: u = x@W1^T, v = x@W3^T (gathered rows), h = silu(c*u)*(c*v) ----
// Block = 8 waves; tile = 32 rows x 256 cols. Each wave: 2 M-tiles x 2 N-tiles
// per weight matrix -> 8 WMMAs per K-step; fragments double-buffered so the
// 12 loads of step k+1 are in flight during step k's WMMAs. `kd` is a runtime
// arg (== kD) so the compiler keeps a compact unroll-2 pipelined loop.

__global__ void __launch_bounds__(256) moe_gemm1(
    const __bf16* __restrict__ Xb, const __bf16* __restrict__ W1,
    const __bf16* __restrict__ W3, const int* __restrict__ cntp,
    const int* __restrict__ idx, const float* __restrict__ ce,
    __bf16* __restrict__ Hb, int kd) {
  const int cnt = *cntp;
  const int m0 = blockIdx.y << 5;
  if (m0 >= cnt) return;  // token-gather: skip tiles past this expert's count

  const int lane = threadIdx.x & 31;
  const int wave = threadIdx.x >> 5;
  const int nb = (blockIdx.x << 8) + (wave << 5);
  const int hi = (lane >> 4) & 1;

  const int r0 = m0 + (lane & 15);
  const int r1 = r0 + 16;
  int arow0 = 0, arow1 = 0;
  if (r0 < cnt) arow0 = idx[r0];
  if (r1 < cnt) arow1 = idx[r1];
  const __bf16* ap0 = Xb + (size_t)arow0 * kD + (hi ? 8 : 0);
  const __bf16* ap1 = Xb + (size_t)arow1 * kD + (hi ? 8 : 0);
  const size_t boff = (size_t)(nb + (lane & 15)) * kD + (hi ? 8 : 0);
  const __bf16* b1p = W1 + boff;
  const __bf16* b3p = W3 + boff;
  constexpr size_t kTile1 = (size_t)16 * kD;  // 16 rows ahead (n+16)

  v8f u00 = {}, u01 = {}, u10 = {}, u11 = {};
  v8f v00 = {}, v01 = {}, v10 = {}, v11 = {};

  // prologue: fragments for k = 0
  v16bf a0  = load_a_frag(ap0);
  v16bf a1  = load_a_frag(ap1);
  v16bf b10 = load_b_frag(b1p);
  v16bf b11 = load_b_frag(b1p + kTile1);
  v16bf b30 = load_b_frag(b3p);
  v16bf b31 = load_b_frag(b3p + kTile1);

#pragma unroll 2
  for (int k = 0; k < kd - 32; k += 32) {
    const int kn = k + 32;
    const v16bf na0  = load_a_frag(ap0 + kn);
    const v16bf na1  = load_a_frag(ap1 + kn);
    const v16bf nb10 = load_b_frag(b1p + kn);
    const v16bf nb11 = load_b_frag(b1p + kTile1 + kn);
    const v16bf nb30 = load_b_frag(b3p + kn);
    const v16bf nb31 = load_b_frag(b3p + kTile1 + kn);

    u00 = wmma_bf16(a0, b10, u00);
    u10 = wmma_bf16(a1, b10, u10);
    u01 = wmma_bf16(a0, b11, u01);
    u11 = wmma_bf16(a1, b11, u11);
    v00 = wmma_bf16(a0, b30, v00);
    v10 = wmma_bf16(a1, b30, v10);
    v01 = wmma_bf16(a0, b31, v01);
    v11 = wmma_bf16(a1, b31, v11);

    a0 = na0; a1 = na1;
    b10 = nb10; b11 = nb11; b30 = nb30; b31 = nb31;
  }
  // epilogue K-step
  u00 = wmma_bf16(a0, b10, u00);
  u10 = wmma_bf16(a1, b10, u10);
  u01 = wmma_bf16(a0, b11, u01);
  u11 = wmma_bf16(a1, b11, u11);
  v00 = wmma_bf16(a0, b30, v00);
  v10 = wmma_bf16(a1, b30, v10);
  v01 = wmma_bf16(a0, b31, v01);
  v11 = wmma_bf16(a1, b31, v11);

  __shared__ float sce[32];
  if (threadIdx.x < 32)
    sce[threadIdx.x] = (m0 + (int)threadIdx.x < cnt) ? ce[m0 + threadIdx.x] : 0.0f;
  __syncthreads();

  const int col = nb + (lane & 15);
  const int rb = hi ? 8 : 0;
#pragma unroll
  for (int mi = 0; mi < 2; ++mi) {
    const v8f& un0 = mi ? u10 : u00;
    const v8f& un1 = mi ? u11 : u01;
    const v8f& vn0 = mi ? v10 : v00;
    const v8f& vn1 = mi ? v11 : v01;
#pragma unroll
    for (int rr = 0; rr < 8; ++rr) {
      const float c = sce[16 * mi + rb + rr];  // 0 for padding rows -> h = 0
      const size_t row = (size_t)(m0 + 16 * mi + rb + rr);
      float uu = c * un0[rr], vv = c * vn0[rr];
      Hb[row * kH + col] = (__bf16)((uu / (1.0f + __expf(-uu))) * vv);
      uu = c * un1[rr];
      vv = c * vn1[rr];
      Hb[row * kH + col + 16] = (__bf16)((uu / (1.0f + __expf(-uu))) * vv);
    }
  }
}

// ---- GEMM2: out[idx[r], :] += h[r, :] @ W2^T (experts run sequentially) ----

__global__ void __launch_bounds__(256) moe_gemm2(
    const __bf16* __restrict__ Hb, const __bf16* __restrict__ W2,
    const int* __restrict__ cntp, const int* __restrict__ idx,
    float* __restrict__ Out, int kh) {
  const int cnt = *cntp;
  const int m0 = blockIdx.y << 5;
  if (m0 >= cnt) return;

  const int lane = threadIdx.x & 31;
  const int wave = threadIdx.x >> 5;
  const int nb = (blockIdx.x << 8) + (wave << 5);
  const int hi = (lane >> 4) & 1;

  const __bf16* ap0 = Hb + (size_t)(m0 + (lane & 15)) * kH + (hi ? 8 : 0);
  const __bf16* ap1 = ap0 + (size_t)16 * kH;
  const __bf16* bp = W2 + (size_t)(nb + (lane & 15)) * kH + (hi ? 8 : 0);
  constexpr size_t kTile1 = (size_t)16 * kH;

  v8f a00 = {}, a01 = {}, a10 = {}, a11 = {};

  v16bf a0 = load_a_frag(ap0);
  v16bf a1 = load_a_frag(ap1);
  v16bf b0 = load_b_frag(bp);
  v16bf b1 = load_b_frag(bp + kTile1);

#pragma unroll 2
  for (int k = 0; k < kh - 32; k += 32) {
    const int kn = k + 32;
    const v16bf na0 = load_a_frag(ap0 + kn);
    const v16bf na1 = load_a_frag(ap1 + kn);
    const v16bf nb0 = load_b_frag(bp + kn);
    const v16bf nb1 = load_b_frag(bp + kTile1 + kn);

    a00 = wmma_bf16(a0, b0, a00);
    a10 = wmma_bf16(a1, b0, a10);
    a01 = wmma_bf16(a0, b1, a01);
    a11 = wmma_bf16(a1, b1, a11);

    a0 = na0; a1 = na1; b0 = nb0; b1 = nb1;
  }
  a00 = wmma_bf16(a0, b0, a00);
  a10 = wmma_bf16(a1, b0, a10);
  a01 = wmma_bf16(a0, b1, a01);
  a11 = wmma_bf16(a1, b1, a11);

  __shared__ int sidx[32];
  if (threadIdx.x < 32)
    sidx[threadIdx.x] = (m0 + (int)threadIdx.x < cnt) ? idx[m0 + threadIdx.x] : 0;
  __syncthreads();

  const int col = nb + (lane & 15);
  const int rb = hi ? 8 : 0;
#pragma unroll
  for (int mi = 0; mi < 2; ++mi) {
    const v8f& an0 = mi ? a10 : a00;
    const v8f& an1 = mi ? a11 : a01;
#pragma unroll
    for (int rr = 0; rr < 8; ++rr) {
      const int r = m0 + 16 * mi + rb + rr;
      if (r < cnt) {  // guard padding rows of the last tile
        float* o = Out + (size_t)sidx[16 * mi + rb + rr] * kD + col;
        o[0]  += an0[rr];
        o[16] += an1[rr];
      }
    }
  }
}

// ---- host orchestration ----

extern "C" void kernel_launch(void* const* d_in, const int* in_sizes, int n_in,
                              void* d_out, int out_size, void* d_ws, size_t ws_size,
                              hipStream_t stream) {
  (void)in_sizes; (void)n_in; (void)out_size; (void)ws_size;
  const float* x  = (const float*)d_in[0];
  const float* gw = (const float*)d_in[1];
  const float* w1 = (const float*)d_in[2];
  const float* w2 = (const float*)d_in[3];
  const float* w3 = (const float*)d_in[4];
  float* out = (float*)d_out;

  // workspace layout (~89 MB):
  //   counts[E] | idx[E*T] | ce[E*T] | x_bf16[T*D] | h_bf16[T*H]
  //   | wb1[H*D] | wb3[H*D] | wb2[D*H]   (per-expert bf16 weights, reused)
  char* ws = (char*)d_ws;
  int*    counts = (int*)ws;
  int*    idx    = (int*)(ws + 256);
  float*  ce     = (float*)(ws + 256 + (size_t)kE * kT * 4);
  __bf16* xb     = (__bf16*)(ws + 256 + (size_t)kE * kT * 8);
  __bf16* hb     = xb  + (size_t)kT * kD;
  __bf16* wb1    = hb  + (size_t)kT * kH;
  __bf16* wb3    = wb1 + (size_t)kH * kD;
  __bf16* wb2    = wb3 + (size_t)kH * kD;

  moe_zero_i32<<<1, 32, 0, stream>>>(counts, kE);
  moe_zero_f32<<<(kT * kD) / 256, 256, 0, stream>>>(out, kT * kD);
  moe_cvt_bf16<<<(kT * kD / 4) / 256, 256, 0, stream>>>(x, xb, kT * kD);
  moe_router<<<kT / 8, 256, 0, stream>>>(x, gw, counts, idx, ce);

  for (int e = 0; e < kE; ++e) {
    // per-expert fp32 -> bf16 weight conversion (stream-ordered, reused buffers)
    moe_cvt_bf16<<<(kH * kD / 4) / 256, 256, 0, stream>>>(
        w1 + (size_t)e * kH * kD, wb1, kH * kD);
    moe_cvt_bf16<<<(kH * kD / 4) / 256, 256, 0, stream>>>(
        w3 + (size_t)e * kH * kD, wb3, kH * kD);
    moe_cvt_bf16<<<(kD * kH / 4) / 256, 256, 0, stream>>>(
        w2 + (size_t)e * kD * kH, wb2, kD * kH);

    moe_gemm1<<<dim3(kH / 256, kT / 32), 256, 0, stream>>>(
        xb, wb1, wb3, counts + e, idx + (size_t)e * kT, ce + (size_t)e * kT, hb,
        kD);
    moe_gemm2<<<dim3(kD / 256, kT / 32), 256, 0, stream>>>(
        hb, wb2, counts + e, idx + (size_t)e * kT, out, kH);
  }
}
